// Block_31791347925466
// MI455X (gfx1250) — compile-verified, matrix operand order checked by reading, not run on hardware
//
#include <hip/hip_runtime.h>
#include <math.h>

typedef __attribute__((ext_vector_type(16))) _Float16 v16h;
typedef __attribute__((ext_vector_type(8)))  float    v8f;

#define WMMA_F16(a, b, c) \
  __builtin_amdgcn_wmma_f32_16x16x32_f16(false, (a), false, (b), (short)0, (c), false, false)

struct Frag {
  union { v16h v; _Float16 h[16]; unsigned int u[8]; };
};

// dword index (k/2) inside a 32-half K-chunk for the 16-bit A/B fragment layout:
// lane%16 = row(A)/col(B), lane/16 selects K groups {8..15, 24..31}
__device__ __forceinline__ int kdw(int kg, int p) {
  return ((p & 4) << 1) | (kg << 2) | (p & 3);
}
// half index base (k) for the same mapping
__device__ __forceinline__ int kbase(int kg, int p) {
  return ((p & 4) << 2) | (kg << 3) | ((p & 3) << 1);
}

// async copy of one 16B chunk global->LDS (ASYNCcnt-tracked, no VGPR data)
__device__ __forceinline__ void async_copy_b128(unsigned lds_off, const void* gptr) {
  asm volatile("global_load_async_to_lds_b128 %0, %1, off"
               :: "v"(lds_off), "v"(gptr) : "memory");
}
__device__ __forceinline__ void wait_async0() {
  asm volatile("s_wait_asynccnt 0x0" ::: "memory");
}

// ---------------- f32 -> f16 convert + transpose: d[n*K+k] = (f16)s[k*N+n] ----------------
__global__ void k_convert_transpose(const float* __restrict__ s, _Float16* __restrict__ d,
                                    int K, int N) {
  int idx = blockIdx.x * 256 + threadIdx.x;
  int total = K * N;
  int stride = gridDim.x * 256;
  for (; idx < total; idx += stride) {
    int k = idx / N, n = idx - k * N;
    d[(size_t)n * K + k] = (_Float16)s[idx];
  }
}

// ---------------- LayerNorm over C=768, one token per block ----------------
__global__ __launch_bounds__(256) void k_layernorm(const float* __restrict__ x,
                                                   const float* __restrict__ w,
                                                   const float* __restrict__ b,
                                                   _Float16* __restrict__ out) {
  const int C = 768;
  __shared__ float red[256];
  const int row = blockIdx.x;
  const int t = threadIdx.x;
  const float* xr = x + (size_t)row * C;
  float v0 = xr[t], v1 = xr[t + 256], v2 = xr[t + 512];
  red[t] = v0 + v1 + v2;
  __syncthreads();
  for (int o = 128; o > 0; o >>= 1) { if (t < o) red[t] += red[t + o]; __syncthreads(); }
  float mu = red[0] * (1.0f / C);
  __syncthreads();
  float d0 = v0 - mu, d1 = v1 - mu, d2 = v2 - mu;
  red[t] = d0 * d0 + d1 * d1 + d2 * d2;
  __syncthreads();
  for (int o = 128; o > 0; o >>= 1) { if (t < o) red[t] += red[t + o]; __syncthreads(); }
  float rs = rsqrtf(red[0] * (1.0f / C) + 1e-5f);
  _Float16* orow = out + (size_t)row * C;
  orow[t]       = (_Float16)(d0 * rs * w[t]       + b[t]);
  orow[t + 256] = (_Float16)(d1 * rs * w[t + 256] + b[t + 256]);
  orow[t + 512] = (_Float16)(d2 * rs * w[t + 512] + b[t + 512]);
}

// ---------------- relative position bias: out[bh,h,w,k] = sum_d q[bh,(h,w),d]*rel[idx,d] ----
__global__ void k_relbias(const _Float16* __restrict__ Q, const float* __restrict__ rel,
                          float* __restrict__ out, int isH) {
  int idx = blockIdx.x * 256 + threadIdx.x;     // 48*32*32*32 total
  int k  = idx & 31;
  int w  = (idx >> 5) & 31;
  int h  = (idx >> 10) & 31;
  int bh = idx >> 15;
  int r = (isH ? h : w) - k + 31;               // 0..62
  const _Float16* qp = Q + (size_t)(bh * 1024 + h * 32 + w) * 64;
  const float* rp = rel + (size_t)r * 64;
  float s = 0.f;
#pragma unroll
  for (int d = 0; d < 64; ++d) s += (float)qp[d] * rp[d];
  out[idx] = s;
}

// ---------------- f16 WMMA GEMM, 64x64 tile, 8 waves, async double-buffered LDS ------------
// A: [M][K] f16 row-major.  Bt: [N][K] f16 (pre-transposed weights).
// mode 0: +bias, scatter to q/k [48][1024][64], v transposed [48][64][1024]   (QKV)
// mode 1: +bias +residual(f32) -> f32 out                                     (proj / mlp2)
// mode 2: +bias, exact GELU -> f16 out                                        (mlp1)
__global__ __launch_bounds__(256) void k_gemm(
    const _Float16* __restrict__ A, const _Float16* __restrict__ Bt,
    int M, int N, int K,
    const float* __restrict__ bias, const float* __restrict__ res,
    float* __restrict__ outf, _Float16* __restrict__ outh,
    _Float16* __restrict__ qh, _Float16* __restrict__ kh, _Float16* __restrict__ vh,
    int mode) {
  __shared__ _Float16 As[2][64 * 32];   // [m][k]
  __shared__ _Float16 Bs[2][64 * 32];   // [n][k]
  const int tid = threadIdx.x;
  const int lane = tid & 31, l16 = lane & 15, kg = lane >> 4;
  const int wid = tid >> 5, wm = wid & 3, wn = wid >> 2;
  const int row0 = blockIdx.y * 64, col0 = blockIdx.x * 64;
  const int t4r = tid >> 2, t4c = tid & 3;            // 64 rows x 4 chunks of 16B

  v8f acc0 = {}, acc1 = {};

  auto stage = [&](int buf, int kc) {
    const _Float16* ga = A  + (size_t)(row0 + t4r) * K + kc + t4c * 8;
    const _Float16* gb = Bt + (size_t)(col0 + t4r) * K + kc + t4c * 8;
    unsigned la = (unsigned)(uintptr_t)(&As[buf][0]) + tid * 16;
    unsigned lb = (unsigned)(uintptr_t)(&Bs[buf][0]) + tid * 16;
    async_copy_b128(la, ga);
    async_copy_b128(lb, gb);
  };

  stage(0, 0);
  int cur = 0;
  for (int kc = 0; kc < K; kc += 32) {
    wait_async0();          // this wave's async writes to As[cur]/Bs[cur] landed
    __syncthreads();        // -> all waves' writes landed; prev readers of cur^1 done
    if (kc + 32 < K) stage(cur ^ 1, kc + 32);

    const unsigned int* Asc = (const unsigned int*)&As[cur][0];
    const unsigned int* Bsu = (const unsigned int*)&Bs[cur][0];
    Frag a, b0, b1;
#pragma unroll
    for (int p = 0; p < 8; ++p) {
      int kd = kdw(kg, p);
      a.u[p]  = Asc[(wm * 16 + l16) * 16 + kd];
      b0.u[p] = Bsu[(wn * 32 + l16) * 16 + kd];
      b1.u[p] = Bsu[(wn * 32 + 16 + l16) * 16 + kd];
    }
    acc0 = WMMA_F16(a.v, b0.v, acc0);
    acc1 = WMMA_F16(a.v, b1.v, acc1);
    cur ^= 1;
  }

#pragma unroll
  for (int t = 0; t < 2; ++t) {
    v8f acc = t ? acc1 : acc0;
#pragma unroll
    for (int r = 0; r < 8; ++r) {
      int gr = row0 + wm * 16 + r + 8 * kg;
      int gc = col0 + wn * 32 + t * 16 + l16;
      float val = acc[r] + bias[gc];
      if (mode == 0) {
        int bb = gr >> 10, s = gr & 1023;
        int which = gc / 768, cc = gc - which * 768;
        int head = cc >> 6, d = cc & 63;
        int bhh = bb * 12 + head;
        if (which == 0)      qh[(size_t)(bhh * 1024 + s) * 64 + d] = (_Float16)val;
        else if (which == 1) kh[(size_t)(bhh * 1024 + s) * 64 + d] = (_Float16)val;
        else                 vh[(size_t)(bhh * 64 + d) * 1024 + s] = (_Float16)val; // V^T
      } else if (mode == 1) {
        outf[(size_t)gr * N + gc] = val + res[(size_t)gr * N + gc];
      } else {
        float g = 0.5f * val * (1.0f + erff(val * 0.70710678118654752f));
        outh[(size_t)gr * N + gc] = (_Float16)g;
      }
    }
  }
}

// ---------------- fused attention: one (head, 16-query tile) per block ----------------
// scores (16x1024) kept in LDS f32; softmax in f32; PV via WMMA with 1/rowsum folded.
// V is stored transposed per head: [48][64][1024] so PV B-fragments are contiguous b128s.
__global__ __launch_bounds__(256) void k_attn(
    const _Float16* __restrict__ Q, const _Float16* __restrict__ Kh,
    const _Float16* __restrict__ Vt, const float* __restrict__ bias_h,
    const float* __restrict__ bias_w, _Float16* __restrict__ out) {
  __shared__ float Sc[16 * 1024];       // 64KB score panel (gfx1250 LDS = 320KB/WGP)
  __shared__ float bhl[16 * 32];
  __shared__ float bwl[16 * 32];
  __shared__ float red[16 * 17];
  __shared__ float rowmax[16];
  __shared__ float rowsum[16];

  const int tid = threadIdx.x;
  const int lane = tid & 31, l16 = lane & 15, kg = lane >> 4;
  const int wid = tid >> 5;
  const int qblk = blockIdx.x, bh = blockIdx.y;
  const int q0 = qblk * 16;
  const int hq = q0 >> 5, w0 = q0 & 31;           // 16 rows never straddle an h boundary
  const _Float16* Qh = Q  + (size_t)bh * 1024 * 64;
  const _Float16* Kp = Kh + (size_t)bh * 1024 * 64;
  const _Float16* Vp = Vt + (size_t)bh * 64 * 1024;

  // stage the 16x32 bias rows for this query tile
  for (int i = tid; i < 512; i += 256) {
    int r = i >> 5, kk = i & 31;
    size_t base = (size_t)((bh * 32 + hq) * 32 + (w0 + r)) * 32 + kk;
    bhl[i] = bias_h[base];
    bwl[i] = bias_w[base];
  }

  // ---- phase 1: S = (Q Kt) * scale, each wave owns 8 of the 64 key tiles ----
  Frag a0, a1;
  const unsigned int* Qdw = (const unsigned int*)Qh;
  const unsigned int* Kdw = (const unsigned int*)Kp;
#pragma unroll
  for (int p = 0; p < 8; ++p) {
    int kd = kdw(kg, p);
    a0.u[p] = Qdw[(size_t)(q0 + l16) * 32 + kd];        // hd 0..31
    a1.u[p] = Qdw[(size_t)(q0 + l16) * 32 + 16 + kd];   // hd 32..63
  }
  for (int tt = 0; tt < 8; ++tt) {
    int col0 = (wid * 8 + tt) * 16;
    Frag b0, b1;
#pragma unroll
    for (int p = 0; p < 8; ++p) {
      int kd = kdw(kg, p);
      b0.u[p] = Kdw[(size_t)(col0 + l16) * 32 + kd];
      b1.u[p] = Kdw[(size_t)(col0 + l16) * 32 + 16 + kd];
    }
    v8f acc = {};
    acc = WMMA_F16(a0.v, b0.v, acc);
    acc = WMMA_F16(a1.v, b1.v, acc);
#pragma unroll
    for (int r = 0; r < 8; ++r) {
      int m = r + 8 * kg;
      Sc[m * 1024 + col0 + l16] = acc[r] * 0.125f;      // scale = hd^-0.5 = 2^-3
    }
  }
  __syncthreads();

  // ---- phase 2: bias add + softmax, 16 threads per row ----
  {
    int r = tid >> 4, c0 = tid & 15;
    float mx = -3.0e38f;
    for (int c = c0; c < 1024; c += 16) {
      float v = Sc[r * 1024 + c] + bhl[r * 32 + (c >> 5)] + bwl[r * 32 + (c & 31)];
      Sc[r * 1024 + c] = v;
      mx = fmaxf(mx, v);
    }
    red[r * 17 + c0] = mx;
    __syncthreads();
    if (tid < 16) {
      float m = red[tid * 17];
      for (int i = 1; i < 16; ++i) m = fmaxf(m, red[tid * 17 + i]);
      rowmax[tid] = m;
    }
    __syncthreads();
    float rm = rowmax[r];
    float sm = 0.f;
    for (int c = c0; c < 1024; c += 16) {
      float e = expf(Sc[r * 1024 + c] - rm);
      Sc[r * 1024 + c] = e;
      sm += e;
    }
    red[r * 17 + c0] = sm;
    __syncthreads();
    if (tid < 16) {
      float s = 0.f;
      for (int i = 0; i < 16; ++i) s += red[tid * 17 + i];
      rowsum[tid] = s;
    }
    __syncthreads();
  }

  // ---- phase 3: O = P V, waves 0-3 each own one 16-col slice of hd ----
  if (wid < 4) {
    int n0 = wid * 16;
    float inv = 1.0f / rowsum[l16];
    const unsigned int* Vdw = (const unsigned int*)Vp;   // rows of V^T: 1024 halves = 512 dw
    v8f acc = {};
    for (int kc = 0; kc < 1024; kc += 32) {
      Frag a, b;
#pragma unroll
      for (int p = 0; p < 8; ++p) {
        int kk = kc + kbase(kg, p);
        a.h[2 * p]     = (_Float16)(Sc[l16 * 1024 + kk] * inv);
        a.h[2 * p + 1] = (_Float16)(Sc[l16 * 1024 + kk + 1] * inv);
        b.u[p] = Vdw[(size_t)(n0 + l16) * 512 + (kc >> 1) + kdw(kg, p)];
      }
      acc = WMMA_F16(a.v, b.v, acc);
    }
    int bb = bh / 12, hd_i = bh % 12;
#pragma unroll
    for (int r = 0; r < 8; ++r) {
      int m = r + 8 * kg;
      out[(size_t)(bb * 1024 + q0 + m) * 768 + hd_i * 64 + n0 + l16] = (_Float16)acc[r];
    }
  }
}

// ---------------- host launcher ----------------
extern "C" void kernel_launch(void* const* d_in, const int* in_sizes, int n_in,
                              void* d_out, int out_size, void* d_ws, size_t ws_size,
                              hipStream_t stream) {
  (void)in_sizes; (void)n_in; (void)out_size; (void)ws_size;
  const float* x      = (const float*)d_in[0];
  const float* n1w    = (const float*)d_in[1];
  const float* n1b    = (const float*)d_in[2];
  const float* qkv_w  = (const float*)d_in[3];
  const float* qkv_b  = (const float*)d_in[4];
  const float* proj_w = (const float*)d_in[5];
  const float* proj_b = (const float*)d_in[6];
  const float* rel_h  = (const float*)d_in[7];
  const float* rel_w  = (const float*)d_in[8];
  const float* n2w    = (const float*)d_in[9];
  const float* n2b    = (const float*)d_in[10];
  const float* mlp_w1 = (const float*)d_in[11];
  const float* mlp_b1 = (const float*)d_in[12];
  const float* mlp_w2 = (const float*)d_in[13];
  const float* mlp_b2 = (const float*)d_in[14];
  float* out = (float*)d_out;

  const int M = 4096, C = 768, TC = 2304, HID = 3072;

  char* ws = (char*)d_ws;
  size_t off = 0;
  auto alloc = [&](size_t bytes) -> void* {
    void* p = ws + off;
    off += bytes;
    off = (off + 255) & ~(size_t)255;
    return p;
  };
  _Float16* qkvw_t = (_Float16*)alloc((size_t)C * TC * 2);    // [2304][768]
  _Float16* projw_t = (_Float16*)alloc((size_t)C * C * 2);    // [768][768]
  _Float16* w1_t   = (_Float16*)alloc((size_t)C * HID * 2);   // [3072][768]
  _Float16* w2_t   = (_Float16*)alloc((size_t)HID * C * 2);   // [768][3072]
  _Float16* xn1    = (_Float16*)alloc((size_t)M * C * 2);
  _Float16* qh     = (_Float16*)alloc((size_t)48 * 1024 * 64 * 2);
  _Float16* kh     = (_Float16*)alloc((size_t)48 * 1024 * 64 * 2);
  _Float16* vh     = (_Float16*)alloc((size_t)48 * 64 * 1024 * 2);  // V transposed
  float*    bhb    = (float*)alloc((size_t)48 * 1024 * 32 * 4);
  float*    bwb    = (float*)alloc((size_t)48 * 1024 * 32 * 4);
  _Float16* attno  = (_Float16*)alloc((size_t)M * C * 2);
  float*    x2     = (float*)alloc((size_t)M * C * 4);
  _Float16* xn2    = (_Float16*)alloc((size_t)M * C * 2);
  _Float16* hbuf   = (_Float16*)alloc((size_t)M * HID * 2);

  k_convert_transpose<<<2048, 256, 0, stream>>>(qkv_w, qkvw_t, C, TC);
  k_convert_transpose<<<2048, 256, 0, stream>>>(proj_w, projw_t, C, C);
  k_convert_transpose<<<2048, 256, 0, stream>>>(mlp_w1, w1_t, C, HID);
  k_convert_transpose<<<2048, 256, 0, stream>>>(mlp_w2, w2_t, HID, C);

  k_layernorm<<<M, 256, 0, stream>>>(x, n1w, n1b, xn1);

  k_gemm<<<dim3(TC / 64, M / 64), 256, 0, stream>>>(
      xn1, qkvw_t, M, TC, C, qkv_b, nullptr, nullptr, nullptr, qh, kh, vh, 0);

  k_relbias<<<6144, 256, 0, stream>>>(qh, rel_h, bhb, 1);
  k_relbias<<<6144, 256, 0, stream>>>(qh, rel_w, bwb, 0);

  k_attn<<<dim3(64, 48), 256, 0, stream>>>(qh, kh, vh, bhb, bwb, attno);

  k_gemm<<<dim3(C / 64, M / 64), 256, 0, stream>>>(
      attno, projw_t, M, C, C, proj_b, x, x2, nullptr, nullptr, nullptr, nullptr, 1);

  k_layernorm<<<M, 256, 0, stream>>>(x2, n2w, n2b, xn2);

  k_gemm<<<dim3(HID / 64, M / 64), 256, 0, stream>>>(
      xn2, w1_t, M, HID, C, mlp_b1, nullptr, nullptr, hbuf, nullptr, nullptr, nullptr, 2);

  k_gemm<<<dim3(C / 64, M / 64), 256, 0, stream>>>(
      hbuf, w2_t, M, C, HID, mlp_b2, x2, out, nullptr, nullptr, nullptr, nullptr, 1);
}
